// TensorBase_39393440038984
// MI455X (gfx1250) — compile-verified, matrix operand order checked by reading, not run on hardware
//
#include <hip/hip_runtime.h>
#include <math.h>

typedef _Float16 h16 __attribute__((ext_vector_type(16)));
typedef _Float16 h8v __attribute__((ext_vector_type(8)));
typedef float    f8  __attribute__((ext_vector_type(8)));
typedef float    f4  __attribute__((ext_vector_type(4)));
typedef unsigned int u4 __attribute__((ext_vector_type(4)));

#define R_RAYS 1024
#define NS     440
#define NSP    448          // padded to 28 tiles of 16
#define NTILE  28
#define NEARV  2.0f
#define FARV   6.0f
#define DIST   ((FARV - NEARV) / (float)(NS - 1))
#define DSCALE 25.0f
#define WTHRES 1.0e-4f
#define GP     256          // plane/line grid
#define GA     128          // alpha volume grid
#define DC     16
#define AC     24
#define APPD   27
#define FC     128
#define INMLP  150
#define K1PAD  160
#define WAVES  4
#define NW1F   (8 * 5 * 512)   // W1^T fragments: 8 mtiles x 5 ksteps x 32 lanes x 16
#define NW2F   (8 * 4 * 512)   // W2^T fragments

// ---------------------------------------------------------------- kernel A: density -> alpha
__global__ void k_density(const float* __restrict__ ro, const float* __restrict__ rd,
                          const float* __restrict__ dplane, const float* __restrict__ dline,
                          const float* __restrict__ avol, float* __restrict__ alpha) {
  int idx = blockIdx.x * blockDim.x + threadIdx.x;
  if (idx >= R_RAYS * NSP) return;
  int r = idx / NSP, s = idx % NSP;
  if (s >= NS) { alpha[idx] = 0.f; return; }
  float z = NEARV + s * DIST;
  float ox = ro[r*3+0], oy = ro[r*3+1], oz = ro[r*3+2];
  float dx = rd[r*3+0], dy = rd[r*3+1], dz = rd[r*3+2];
  float px = ox + dx*z, py = oy + dy*z, pz = oz + dz*z;
  bool rv = (px >= -1.5f) && (px <= 1.5f) && (py >= -1.5f) && (py <= 1.5f) &&
            (pz >= -1.5f) && (pz <= 1.5f);
  float xn = (px + 1.5f) * (2.f/3.f) - 1.f;
  float yn = (py + 1.5f) * (2.f/3.f) - 1.f;
  float zn = (pz + 1.5f) * (2.f/3.f) - 1.f;

  float fx = (xn + 1.f) * 0.5f * (GA - 1);
  float fy = (yn + 1.f) * 0.5f * (GA - 1);
  float fz = (zn + 1.f) * 0.5f * (GA - 1);
  int x0 = min(max((int)floorf(fx), 0), GA - 2);
  int y0 = min(max((int)floorf(fy), 0), GA - 2);
  int z0 = min(max((int)floorf(fz), 0), GA - 2);
  float wx = fminf(fmaxf(fx - (float)x0, 0.f), 1.f);
  float wy = fminf(fmaxf(fy - (float)y0, 0.f), 1.f);
  float wz = fminf(fmaxf(fz - (float)z0, 0.f), 1.f);
  const float* v0 = avol + ((size_t)z0 * GA + y0) * GA + x0;
  const float* v1 = v0 + (size_t)GA * GA;
  float c00 = v0[0]  * (1.f-wx) + v0[1]    * wx;
  float c01 = v0[GA] * (1.f-wx) + v0[GA+1] * wx;
  float c10 = v1[0]  * (1.f-wx) + v1[1]    * wx;
  float c11 = v1[GA] * (1.f-wx) + v1[GA+1] * wx;
  float aval = (c00*(1.f-wy) + c01*wy) * (1.f-wz) + (c10*(1.f-wy) + c11*wy) * wz;

  float sigma = 0.f;
  if (rv && aval > 0.f) {
    float us[3] = {xn, xn, yn};
    float vs[3] = {yn, zn, zn};
    float ts[3] = {zn, yn, xn};
    float feat = 0.f;
    #pragma unroll
    for (int p = 0; p < 3; ++p) {
      float pfx = (us[p] + 1.f) * 0.5f * (GP - 1);
      float pfy = (vs[p] + 1.f) * 0.5f * (GP - 1);
      float lft = (ts[p] + 1.f) * 0.5f * (GP - 1);
      int px0 = min(max((int)floorf(pfx), 0), GP - 2);
      int py0 = min(max((int)floorf(pfy), 0), GP - 2);
      int lt0 = min(max((int)floorf(lft), 0), GP - 2);
      float pwx = fminf(fmaxf(pfx - (float)px0, 0.f), 1.f);
      float pwy = fminf(fmaxf(pfy - (float)py0, 0.f), 1.f);
      float lwt = fminf(fmaxf(lft - (float)lt0, 0.f), 1.f);
      float w00 = (1.f-pwx)*(1.f-pwy), w01 = pwx*(1.f-pwy);
      float w10 = (1.f-pwx)*pwy,       w11 = pwx*pwy;
      const float* pb = dplane + (size_t)p * DC * GP * GP;
      const float* lb = dline  + (size_t)p * DC * GP;
      for (int c = 0; c < DC; ++c) {
        const float* pc = pb + (size_t)c * GP * GP + (size_t)py0 * GP + px0;
        float pv = pc[0]*w00 + pc[1]*w01 + pc[GP]*w10 + pc[GP+1]*w11;
        const float* lc = lb + c * GP + lt0;
        float lv = lc[0]*(1.f-lwt) + lc[1]*lwt;
        feat += pv * lv;
      }
    }
    sigma = (feat > 20.f) ? feat : log1pf(__expf(feat));   // softplus
  }
  float dd = (s < NS - 1) ? DIST : 0.f;
  alpha[idx] = 1.f - __expf(-sigma * dd * DSCALE);
}

// ---------------------------------------------------------------- kernel W: weights -> f16 fragment layout
// Transposed formulation: h^T = W^T @ x^T. A-fragment (16x32 slice of W^T) per (mtile,kstep):
//   lane: m = mtile*16 + (lane&15); kb = kstep*32 + (lane>>4)*8
//   a[i] (i<8)  = W^T[m][kb+i]    = W[kb+i][m]
//   a[i] (i>=8) = W^T[m][kb+8+i]  = W[kb+8+i][m]
// stored contiguously: frag*512 + lane*16 + i  -> one aligned 32B load per lane.
__global__ void k_prep(const float* __restrict__ W1, const float* __restrict__ W2,
                       const float* __restrict__ W3,
                       _Float16* __restrict__ w1f, _Float16* __restrict__ w2f,
                       float* __restrict__ w3t) {
  int i = blockIdx.x * blockDim.x + threadIdx.x;
  if (i < NW1F) {
    int e = i & 15, lane = (i >> 4) & 31, frag = i >> 9;
    int kstep = frag >> 3, mtile = frag & 7;
    int m  = mtile * 16 + (lane & 15);
    int kb = kstep * 32 + ((lane >> 4) << 3);
    int row = kb + e + ((e >= 8) ? 8 : 0);
    w1f[i] = (row < INMLP) ? (_Float16)W1[row * FC + m] : (_Float16)0.f;
  } else if (i < NW1F + NW2F) {
    int t = i - NW1F;
    int e = t & 15, lane = (t >> 4) & 31, frag = t >> 9;
    int kstep = frag >> 3, mtile = frag & 7;
    int m  = mtile * 16 + (lane & 15);
    int kb = kstep * 32 + ((lane >> 4) << 3);
    int row = kb + e + ((e >= 8) ? 8 : 0);
    w2f[t] = (_Float16)W2[row * FC + m];
  } else if (i < NW1F + NW2F + 3 * FC) {
    int t = i - NW1F - NW2F;
    int c = t / FC, k = t % FC;
    w3t[t] = W3[k * 3 + c];           // W3^T [3][128]
  }
}

// ---------------------------------------------------------------- kernel B: transmittance scan
__global__ void k_scan(const float* __restrict__ alpha, float* __restrict__ wt,
                       float* __restrict__ acc, float* __restrict__ depth,
                       float* __restrict__ rgbacc) {
  int r = blockIdx.x * blockDim.x + threadIdx.x;
  if (r >= R_RAYS) return;
  float T = 1.f, a = 0.f, dsum = 0.f;
  for (int s = 0; s < NS; ++s) {
    float al = alpha[r * NSP + s];
    float w = al * T;
    T *= (1.f - al + 1e-10f);
    wt[r * NSP + s] = w;
    a += w;
    dsum += w * (NEARV + s * DIST);
  }
  for (int s = NS; s < NSP; ++s) wt[r * NSP + s] = 0.f;
  acc[r] = a;
  depth[r] = dsum;
  rgbacc[r*3+0] = 0.f; rgbacc[r*3+1] = 0.f; rgbacc[r*3+2] = 0.f;
}

// ---------------------------------------------------------------- fragment helpers
__device__ __forceinline__ h16 load_wfrag(const _Float16* wf, int frag, int lane) {
  return *(const h16*)(wf + (frag << 9) + (lane << 4));
}
__device__ __forceinline__ h16 load_act(const _Float16* base, int ldk, int k0, int lane) {
  int n  = lane & 15;
  int kb = k0 + ((lane >> 4) << 3);
  const _Float16* p = base + n * ldk + kb;
  h8v lo = *(const h8v*)p;
  h8v hi = *(const h8v*)(p + 16);
  return __builtin_shufflevector(lo, hi, 0,1,2,3,4,5,6,7,8,9,10,11,12,13,14,15);
}
__device__ __forceinline__ void store_h(_Float16* hout, int ldk, const f8 acc,
                                        const float* bias, int mtile, int lane) {
  int n  = lane & 15;
  int cb = mtile * 16 + ((lane >> 4) << 3);
  f4 b0 = *(const f4*)(bias + cb);
  f4 b1 = *(const f4*)(bias + cb + 4);
  h8v hv;
  #pragma unroll
  for (int r = 0; r < 8; ++r) {
    float bb = (r < 4) ? b0[r] : b1[r - 4];
    float v = acc[r] + bb;
    hv[r] = (_Float16)(v > 0.f ? v : 0.f);
  }
  *(h8v*)(hout + n * ldk + cb) = hv;
}

// ---------------------------------------------------------------- kernel C: MLP via WMMA
__global__ void __launch_bounds__(WAVES * 32) k_mlp(
    const float* __restrict__ ro, const float* __restrict__ rd,
    const float* __restrict__ aplane, const float* __restrict__ aline,
    const float* __restrict__ basis,
    const _Float16* __restrict__ w1f, const float* __restrict__ b1,
    const _Float16* __restrict__ w2f, const float* __restrict__ b2,
    const float* __restrict__ w3t, const float* __restrict__ b3,
    const float* __restrict__ wt, float* __restrict__ rgbacc) {
  __shared__ __align__(32) _Float16 sW1f[NW1F];     // 40 KB
  __shared__ __align__(32) _Float16 sW2f[NW2F];     // 32 KB
  __shared__ __align__(32) float sBasis[72 * APPD];
  __shared__ __align__(32) float sW3T[3 * FC];
  __shared__ __align__(32) float sB1[FC], sB2[FC], sB3[4];
  __shared__ __align__(32) _Float16 sIn[WAVES][16 * K1PAD];  // mlp_in -> h1 -> h2
  __shared__ __align__(32) float sPF[WAVES][16 * 72];
  __shared__ __align__(32) float sAF[WAVES][16 * APPD];
  __shared__ __align__(32) float sRGB[WAVES][16 * 3];

  int t = threadIdx.x;
  // wide cooperative staging of weights
  {
    const u4* s1 = (const u4*)w1f; u4* d1 = (u4*)sW1f;
    for (int i = t; i < NW1F * 2 / 16; i += blockDim.x) d1[i] = s1[i];
    const u4* s2 = (const u4*)w2f; u4* d2 = (u4*)sW2f;
    for (int i = t; i < NW2F * 2 / 16; i += blockDim.x) d2[i] = s2[i];
  }
  for (int i = t; i < 72 * APPD; i += blockDim.x) sBasis[i] = basis[i];
  for (int i = t; i < 3 * FC;   i += blockDim.x) sW3T[i] = w3t[i];
  for (int i = t; i < FC;       i += blockDim.x) { sB1[i] = b1[i]; sB2[i] = b2[i]; }
  if (t < 3) sB3[t] = b3[t];
  __syncthreads();

  int wv = t >> 5, lane = t & 31;
  int tile = blockIdx.x * WAVES + wv;
  int ray  = tile / NTILE;
  int s0   = (tile % NTILE) * 16;

  float w_s = 0.f;
  if (lane < 16) w_s = wt[ray * NSP + s0 + lane];
  float mk = (lane < 16 && w_s > WTHRES) ? 1.f : 0.f;
  #pragma unroll
  for (int off = 8; off; off >>= 1) mk += __shfl_down(mk, off, 32);
  mk = __shfl(mk, 0, 32);
  if (mk > 0.f) {
    _Float16* mIn = sIn[wv];
    float* mPF = sPF[wv];
    float* mAF = sAF[wv];
    float* mRGB = sRGB[wv];

    float dx = rd[ray*3+0], dy = rd[ray*3+1], dz = rd[ray*3+2];
    float ox = ro[ray*3+0], oy = ro[ray*3+1], oz = ro[ray*3+2];

    // --- gather 72 app plane*line products; 2 lanes per sample ---
    {
      int s = lane >> 1, half = lane & 1;
      float z = NEARV + (s0 + s) * DIST;
      float px = ox + dx*z, py = oy + dy*z, pz = oz + dz*z;
      float xn = (px + 1.5f) * (2.f/3.f) - 1.f;
      float yn = (py + 1.5f) * (2.f/3.f) - 1.f;
      float zn = (pz + 1.5f) * (2.f/3.f) - 1.f;
      float us[3] = {xn, xn, yn};
      float vs[3] = {yn, zn, zn};
      float ts[3] = {zn, yn, xn};
      int X0[3], Y0[3], T0[3];
      float WX[3], WY[3], WT[3];
      #pragma unroll
      for (int p = 0; p < 3; ++p) {
        float pfx = (us[p] + 1.f) * 0.5f * (GP - 1);
        float pfy = (vs[p] + 1.f) * 0.5f * (GP - 1);
        float lft = (ts[p] + 1.f) * 0.5f * (GP - 1);
        X0[p] = min(max((int)floorf(pfx), 0), GP - 2);
        Y0[p] = min(max((int)floorf(pfy), 0), GP - 2);
        T0[p] = min(max((int)floorf(lft), 0), GP - 2);
        WX[p] = fminf(fmaxf(pfx - (float)X0[p], 0.f), 1.f);
        WY[p] = fminf(fmaxf(pfy - (float)Y0[p], 0.f), 1.f);
        WT[p] = fminf(fmaxf(lft - (float)T0[p], 0.f), 1.f);
      }
      for (int pc = half * 36; pc < half * 36 + 36; ++pc) {
        int p = pc / AC, c = pc % AC;
        float w00 = (1.f-WX[p])*(1.f-WY[p]), w01 = WX[p]*(1.f-WY[p]);
        float w10 = (1.f-WX[p])*WY[p],       w11 = WX[p]*WY[p];
        const float* pcp = aplane + (((size_t)p * AC + c) * GP + Y0[p]) * GP + X0[p];
        float pv = pcp[0]*w00 + pcp[1]*w01 + pcp[GP]*w10 + pcp[GP+1]*w11;
        const float* lc = aline + ((size_t)p * AC + c) * GP + T0[p];
        float lv = lc[0]*(1.f-WT[p]) + lc[1]*WT[p];
        mPF[s * 72 + pc] = pv * lv;
      }
    }
    // --- app_feat = pf @ basis (16x72 x 72x27) ---
    for (int idx = lane; idx < 16 * APPD; idx += 32) {
      int s = idx / APPD, j = idx % APPD;
      float sum = 0.f;
      for (int k = 0; k < 72; ++k) sum += mPF[s * 72 + k] * sBasis[k * APPD + j];
      mAF[idx] = sum;
    }
    // --- build mlp_in (16 x 160 f16, row-major) ---
    for (int e = lane; e < 16 * K1PAD; e += 32) {
      int s = e / K1PAD, col = e % K1PAD;
      float v = 0.f;
      if (col < APPD)      v = mAF[s * APPD + col];
      else if (col < 30)   v = (col == 27) ? dx : ((col == 28) ? dy : dz);
      else if (col < 84)  { int q = col - 30;  v = __sinf(mAF[s*APPD + (q>>1)] * ((q&1) ? 2.f : 1.f)); }
      else if (col < 138) { int q = col - 84;  v = __cosf(mAF[s*APPD + (q>>1)] * ((q&1) ? 2.f : 1.f)); }
      else if (col < 144) { int q = col - 138; float vd = ((q>>1)==0)?dx:(((q>>1)==1)?dy:dz);
                            v = __sinf(vd * ((q&1) ? 2.f : 1.f)); }
      else if (col < 150) { int q = col - 144; float vd = ((q>>1)==0)?dx:(((q>>1)==1)?dy:dz);
                            v = __cosf(vd * ((q&1) ? 2.f : 1.f)); }
      mIn[s * K1PAD + col] = (_Float16)v;
    }

    f8 accm[8];
    h16 aw[8];
    // ---------------- layer 1: h1^T = W1^T(128x160) @ x^T(160x16) ----------------
    #pragma unroll
    for (int n = 0; n < 8; ++n)
      #pragma unroll
      for (int i = 0; i < 8; ++i) accm[n][i] = 0.f;
    #pragma unroll
    for (int k = 0; k < 5; ++k) {
      // batch: one activation fragment + all 8 weight fragments, then 8 WMMAs
      h16 bx = load_act(mIn, K1PAD, k * 32, lane);
      #pragma unroll
      for (int mt = 0; mt < 8; ++mt) aw[mt] = load_wfrag(sW1f, k * 8 + mt, lane);
      #pragma unroll
      for (int mt = 0; mt < 8; ++mt)
        accm[mt] = __builtin_amdgcn_wmma_f32_16x16x32_f16(false, aw[mt], false, bx,
                                                          (short)0, accm[mt], false, false);
    }
    #pragma unroll
    for (int mt = 0; mt < 8; ++mt) store_h(mIn, K1PAD, accm[mt], sB1, mt, lane);

    // ---------------- layer 2: h2^T = W2^T(128x128) @ h1^T(128x16) ----------------
    #pragma unroll
    for (int n = 0; n < 8; ++n)
      #pragma unroll
      for (int i = 0; i < 8; ++i) accm[n][i] = 0.f;
    #pragma unroll
    for (int k = 0; k < 4; ++k) {
      h16 bx = load_act(mIn, K1PAD, k * 32, lane);
      #pragma unroll
      for (int mt = 0; mt < 8; ++mt) aw[mt] = load_wfrag(sW2f, k * 8 + mt, lane);
      #pragma unroll
      for (int mt = 0; mt < 8; ++mt)
        accm[mt] = __builtin_amdgcn_wmma_f32_16x16x32_f16(false, aw[mt], false, bx,
                                                          (short)0, accm[mt], false, false);
    }
    #pragma unroll
    for (int mt = 0; mt < 8; ++mt) store_h(mIn, K1PAD, accm[mt], sB2, mt, lane);

    // ---------------- layer 3: 16x128 @ 128x3 + sigmoid ----------------
    for (int idx = lane; idx < 48; idx += 32) {
      int s = idx / 3, c = idx % 3;
      float sum = sB3[c];
      const _Float16* hp = mIn + s * K1PAD;
      const float* wp = sW3T + c * FC;
      for (int k = 0; k < FC; k += 8) {
        h8v hv = *(const h8v*)(hp + k);
        f4 w0 = *(const f4*)(wp + k);
        f4 w1v = *(const f4*)(wp + k + 4);
        #pragma unroll
        for (int j = 0; j < 4; ++j) sum += (float)hv[j] * w0[j];
        #pragma unroll
        for (int j = 0; j < 4; ++j) sum += (float)hv[4 + j] * w1v[j];
      }
      mRGB[idx] = 1.f / (1.f + __expf(-sum));
    }
    // ---------------- composite: per-ray atomic accumulation ----------------
    #pragma unroll
    for (int c = 0; c < 3; ++c) {
      float v = 0.f;
      if (lane < 16 && w_s > WTHRES) v = w_s * mRGB[lane * 3 + c];
      #pragma unroll
      for (int off = 8; off; off >>= 1) v += __shfl_down(v, off, 32);
      if (lane == 0) atomicAdd(&rgbacc[ray * 3 + c], v);
    }
  }
}

// ---------------------------------------------------------------- kernel D: finalize
__global__ void k_final(const float* __restrict__ rgbacc, const float* __restrict__ acc,
                        const float* __restrict__ depth, float* __restrict__ out) {
  int r = blockIdx.x * blockDim.x + threadIdx.x;
  if (r >= R_RAYS) return;
  float bg = 1.f - acc[r];
  #pragma unroll
  for (int c = 0; c < 3; ++c) {
    float v = rgbacc[r * 3 + c] + bg;
    out[r * 3 + c] = fminf(fmaxf(v, 0.f), 1.f);
  }
  out[3 * R_RAYS + r] = depth[r];
}

// ---------------------------------------------------------------- launch
extern "C" void kernel_launch(void* const* d_in, const int* in_sizes, int n_in,
                              void* d_out, int out_size, void* d_ws, size_t ws_size,
                              hipStream_t stream) {
  (void)in_sizes; (void)n_in; (void)out_size; (void)ws_size;
  const float* ro     = (const float*)d_in[0];
  const float* rd     = (const float*)d_in[1];
  const float* dplane = (const float*)d_in[2];
  const float* dline  = (const float*)d_in[3];
  const float* aplane = (const float*)d_in[4];
  const float* aline  = (const float*)d_in[5];
  const float* basis  = (const float*)d_in[6];
  const float* W1     = (const float*)d_in[7];
  const float* b1     = (const float*)d_in[8];
  const float* W2     = (const float*)d_in[9];
  const float* b2     = (const float*)d_in[10];
  const float* W3     = (const float*)d_in[11];
  const float* b3     = (const float*)d_in[12];
  const float* avol   = (const float*)d_in[13];
  float* out = (float*)d_out;

  char* ws = (char*)d_ws;
  size_t off = 0;
  float* alpha  = (float*)(ws + off); off += (size_t)R_RAYS * NSP * 4;
  float* wt     = (float*)(ws + off); off += (size_t)R_RAYS * NSP * 4;
  float* acc    = (float*)(ws + off); off += (size_t)R_RAYS * 4;
  float* depth  = (float*)(ws + off); off += (size_t)R_RAYS * 4;
  float* rgbacc = (float*)(ws + off); off += (size_t)R_RAYS * 3 * 4;
  _Float16* w1f = (_Float16*)(ws + off); off += (size_t)NW1F * 2;
  _Float16* w2f = (_Float16*)(ws + off); off += (size_t)NW2F * 2;
  float* w3t    = (float*)(ws + off); off += (size_t)3 * FC * 4;

  k_density<<<(R_RAYS * NSP + 255) / 256, 256, 0, stream>>>(ro, rd, dplane, dline, avol, alpha);
  k_prep<<<(NW1F + NW2F + 3 * FC + 255) / 256, 256, 0, stream>>>(W1, W2, W3, w1f, w2f, w3t);
  k_scan<<<(R_RAYS + 255) / 256, 256, 0, stream>>>(alpha, wt, acc, depth, rgbacc);
  k_mlp<<<(R_RAYS * NTILE) / WAVES, WAVES * 32, 0, stream>>>(
      ro, rd, aplane, aline, basis, w1f, b1, w2f, b2, w3t, b3, wt, rgbacc);
  k_final<<<(R_RAYS + 255) / 256, 256, 0, stream>>>(rgbacc, acc, depth, out);
}
